// Codebook_5574867550330
// MI455X (gfx1250) — matrix operand order repaired and checked by reading, not compiled, not run on hardware
//
#include <hip/hip_runtime.h>
#include <hip/hip_bf16.h>
#include <stdint.h>

// CDNA5 WMMA operand types (wave32): 16 bf16 per lane = 8 VGPRs; 8 f32 acc.
typedef __attribute__((ext_vector_type(16))) __bf16 v16bf;
typedef __attribute__((ext_vector_type(8)))  float  v8f;
typedef __attribute__((ext_vector_type(4)))  int    v4i;

constexpr int DIMS    = 128;            // GEMM reduction dim
constexpr int KCODES  = 2048;           // number of codes (GEMM N dim)
constexpr int MTILE   = 128;            // feature rows per block
constexpr int NCHUNK  = 32;             // codes per LDS buffer (double-buffered)
constexpr int NCHUNKS = KCODES / NCHUNK;
constexpr int LDSW    = 136;            // padded row stride in bf16 elems (272B = 68 dwords
                                        //  -> lane L hits bank 4L: conflict-free b128 frags)

__device__ __forceinline__ unsigned short bf16_rne(float f) {
  unsigned int u = __float_as_uint(f);
  u += 0x7FFFu + ((u >> 16) & 1u);      // round-to-nearest-even
  return (unsigned short)(u >> 16);
}
__device__ __forceinline__ float bf16_to_f32(unsigned short h) {
  return __uint_as_float(((unsigned int)h) << 16);
}

union Frag16 { v16bf v; unsigned short u[16]; uint4 q[2]; };

// ---- CDNA5 async global->LDS copy (ASYNCcnt-tracked, bypasses VGPRs) ----
// Builtin signature (from clang diagnostic): (v4i __device__*, v4i __shared__*, int, int)
#if __has_builtin(__builtin_amdgcn_global_load_async_to_lds_b128)
#define ASYNC_LDS 1
__device__ __forceinline__ void async_cp16(const void* g, void* l) {
  __builtin_amdgcn_global_load_async_to_lds_b128(
      (__attribute__((address_space(1))) v4i*)(uintptr_t)g,   // global AS1 (same bits)
      (__attribute__((address_space(3))) v4i*)l,              // generic->LDS addrspacecast
      0, 0);
}
template <int N> __device__ __forceinline__ void wait_async() {
#if __has_builtin(__builtin_amdgcn_s_wait_asynccnt)
  __builtin_amdgcn_s_wait_asynccnt(N);
#else
  asm volatile("s_wait_asynccnt %0" ::"i"(N) : "memory");
#endif
}
#endif

// --- prep: csq[k] = 0.5 * ||codes[k]||^2 ---------------------------------
__global__ void csq_half_kernel(const float* __restrict__ codes,
                                float* __restrict__ csq) {
  int k = blockIdx.x * blockDim.x + threadIdx.x;
  if (k < KCODES) {
    const float* c = codes + (size_t)k * DIMS;
    float s = 0.f;
#pragma unroll 8
    for (int d = 0; d < DIMS; ++d) { float v = c[d]; s += v * v; }
    csq[k] = 0.5f * s;
  }
}

// --- prep: split codebook once into bf16 hi/lo (codes ~ hi + lo) ---------
__global__ void split_codes_kernel(const float* __restrict__ codes,
                                   unsigned short* __restrict__ chi,
                                   unsigned short* __restrict__ clo) {
  int e4 = blockIdx.x * blockDim.x + threadIdx.x;   // float4 index
  if (e4 < KCODES * DIMS / 4) {
    float4 v = ((const float4*)codes)[e4];
    float cv[4] = {v.x, v.y, v.z, v.w};
    unsigned int h[4], l[4];
#pragma unroll
    for (int j = 0; j < 4; ++j) {
      h[j] = bf16_rne(cv[j]);
      l[j] = bf16_rne(cv[j] - bf16_to_f32((unsigned short)h[j]));
    }
    uint2 ph = {h[0] | (h[1] << 16), h[2] | (h[3] << 16)};
    uint2 pl = {l[0] | (l[1] << 16), l[2] | (l[3] << 16)};
    ((uint2*)chi)[e4] = ph;
    ((uint2*)clo)[e4] = pl;
  }
}

// stage one 32-code chunk of pre-split bf16 codes into a padded LDS buffer.
// 32 rows x 16 b128 segs per (hi|lo) = 1024 b128 xfers; 256 threads -> 4 each.
__device__ __forceinline__ void stage_chunk(const unsigned short* __restrict__ chi,
                                            const unsigned short* __restrict__ clo,
                                            unsigned short* sHi, unsigned short* sLo,
                                            int chunk, int tid) {
#pragma unroll
  for (int i = 0; i < 2; ++i) {
    const int q   = tid + i * 256;                 // 0..511
    const int row = q >> 4;
    const int seg = q & 15;
    const size_t goff = (size_t)(chunk * NCHUNK + row) * DIMS + seg * 8;  // ushort idx
    const int    loff = row * LDSW + seg * 8;
#ifdef ASYNC_LDS
    async_cp16(chi + goff, sHi + loff);
    async_cp16(clo + goff, sLo + loff);
#else
    *(uint4*)(sHi + loff) = *(const uint4*)(chi + goff);
    *(uint4*)(sLo + loff) = *(const uint4*)(clo + goff);
#endif
  }
}

// --- main: double-buffered split-bf16 WMMA GEMM + argmin + gather --------
__global__ __launch_bounds__(256) void vq_main_kernel(
    const float* __restrict__ features,
    const float* __restrict__ codes,          // f32, for the final gather
    const unsigned short* __restrict__ chi,   // pre-split bf16 hi
    const unsigned short* __restrict__ clo,   // pre-split bf16 lo
    const float* __restrict__ csq,            // 0.5*||c||^2
    float* __restrict__ out)
{
  __shared__ unsigned short sB[2][2][NCHUNK * LDSW];   // [buf][hi|lo], ~35 KB
  __shared__ int sIdx[MTILE];

  const int tid  = threadIdx.x;
  const int wave = tid >> 5;
  const int lane = tid & 31;
  const int half = lane >> 4;
  const int l16  = lane & 15;

  // ---- A tile: this wave's 16 rows x 128 dims, f32 -> bf16 hi/lo in regs.
  // ISA 16-bit A layout (16x32): lane%16 -> row M; lanes 0-15 hold K runs
  // {k0..k0+7} (VGPR0-3) and {k0+16..k0+23} (VGPR4-7); lanes 16-31 hold +8.
  const int m_row = blockIdx.x * MTILE + wave * 16 + l16;
  const float* frow = features + (size_t)m_row * DIMS;
  Frag16 a_hi[4], a_lo[4];
#pragma unroll
  for (int ks = 0; ks < 4; ++ks) {
    const int b1 = ks * 32 + half * 8;
#pragma unroll
    for (int run = 0; run < 2; ++run) {
      const int kb = b1 + run * 16;
      float4 f0 = *(const float4*)(frow + kb);
      float4 f1 = *(const float4*)(frow + kb + 4);
      float fv[8] = {f0.x, f0.y, f0.z, f0.w, f1.x, f1.y, f1.z, f1.w};
#pragma unroll
      for (int j = 0; j < 8; ++j) {
        unsigned short h = bf16_rne(fv[j]);
        a_hi[ks].u[run * 8 + j] = h;
        a_lo[ks].u[run * 8 + j] = bf16_rne(fv[j] - bf16_to_f32(h));
      }
    }
  }

  float run_val[8]; int run_n0[8];
#pragma unroll
  for (int r = 0; r < 8; ++r) { run_val[r] = -3.4e38f; run_n0[r] = 0; }

  // prologue: kick off chunk 0 into buffer 0
  stage_chunk(chi, clo, sB[0][0], sB[0][1], 0, tid);

  for (int i = 0; i < NCHUNKS; ++i) {
    const int buf = i & 1;
    if (i + 1 < NCHUNKS) {
      // issue next chunk into the other buffer (its prior data was consumed
      // before the end-of-iteration barrier of i-1)
      stage_chunk(chi, clo, sB[buf ^ 1][0], sB[buf ^ 1][1], i + 1, tid);
#ifdef ASYNC_LDS
      wait_async<4>();   // in-order: chunk i's 4 xfers retired, next 4 in flight
#endif
    } else {
#ifdef ASYNC_LDS
      wait_async<0>();
#endif
    }
    __syncthreads();     // all waves' chunk-i data visible in LDS

    const unsigned short* sHi = sB[buf][0];
    const unsigned short* sLo = sB[buf][1];
#pragma unroll
    for (int sub = 0; sub < NCHUNK / 16; ++sub) {
      const int n0 = i * NCHUNK + sub * 16;
      v8f acc = {};
#pragma unroll
      for (int ks = 0; ks < 4; ++ks) {
        // ISA 16-bit B layout (32x16): lane%16 -> column N (= code row),
        // lanes 0-15 hold K runs {k0..k0+7},{k0+16..k0+23}; lanes 16-31 hold +8.
        const int cr   = sub * 16 + l16;
        const int koff = ks * 32 + half * 8;
        Frag16 b_hi, b_lo;
        const unsigned short* ph = &sHi[cr * LDSW + koff];
        b_hi.q[0] = *(const uint4*)(ph);
        b_hi.q[1] = *(const uint4*)(ph + 16);
        const unsigned short* pl = &sLo[cr * LDSW + koff];
        b_lo.q[0] = *(const uint4*)(pl);
        b_lo.q[1] = *(const uint4*)(pl + 16);
        // split product: hi*hi + hi*lo + lo*hi  (lo*lo ~2^-18, dropped)
        acc = __builtin_amdgcn_wmma_f32_16x16x32_bf16(false, a_hi[ks].v, false, b_hi.v,
                                                      (short)0, acc, false, false);
        acc = __builtin_amdgcn_wmma_f32_16x16x32_bf16(false, a_hi[ks].v, false, b_lo.v,
                                                      (short)0, acc, false, false);
        acc = __builtin_amdgcn_wmma_f32_16x16x32_bf16(false, a_lo[ks].v, false, b_hi.v,
                                                      (short)0, acc, false, false);
      }
      // C/D layout: VGPR r, lanes0-15 -> M=r, lanes16-31 -> M=8+r; N=lane%16
      const float bias = csq[n0 + l16];
#pragma unroll
      for (int r = 0; r < 8; ++r) {
        float s = acc[r] - bias;
        if (s > run_val[r]) { run_val[r] = s; run_n0[r] = n0; }
      }
    }
    __syncthreads();     // protect buf from being overwritten by i+1's stage
  }

  // ---- per-row argmax across the 16 lanes of each half (first-idx ties) ----
#pragma unroll
  for (int r = 0; r < 8; ++r) {
    float v = run_val[r];
    int   i = run_n0[r] + l16;
#pragma unroll
    for (int off = 8; off > 0; off >>= 1) {
      float ov = __shfl_xor(v, off, 16);
      int   oi = __shfl_xor(i, off, 16);
      if (ov > v || (ov == v && oi < i)) { v = ov; i = oi; }
    }
    if (l16 == 0) sIdx[wave * 16 + half * 8 + r] = i;
  }
  __syncthreads();

  // ---- out[row] = codes[idx[row]]  (coalesced float4 gather) ----
  for (int e = tid; e < MTILE * (DIMS / 4); e += 256) {
    const int row = e >> 5;
    const int c4  = e & 31;
    const int k   = sIdx[row];
    float4 v = ((const float4*)(codes + (size_t)k * DIMS))[c4];
    ((float4*)(out + ((size_t)(blockIdx.x * MTILE + row)) * DIMS))[c4] = v;
  }
}

// --- fallback (small d_ws): self-contained, converts codes per block -----
__global__ __launch_bounds__(256) void vq_fallback_kernel(
    const float* __restrict__ features,
    const float* __restrict__ codes,
    const float* __restrict__ csq,
    float* __restrict__ out)
{
  constexpr int NC = 64;
  __shared__ unsigned short sBhi[NC * LDSW];
  __shared__ unsigned short sBlo[NC * LDSW];
  __shared__ int sIdx[MTILE];

  const int tid  = threadIdx.x;
  const int wave = tid >> 5;
  const int lane = tid & 31;
  const int half = lane >> 4;
  const int l16  = lane & 15;

  const int m_row = blockIdx.x * MTILE + wave * 16 + l16;
  const float* frow = features + (size_t)m_row * DIMS;
  Frag16 a_hi[4], a_lo[4];
#pragma unroll
  for (int ks = 0; ks < 4; ++ks) {
    const int b1 = ks * 32 + half * 8;
#pragma unroll
    for (int run = 0; run < 2; ++run) {
      const int kb = b1 + run * 16;
      float4 f0 = *(const float4*)(frow + kb);
      float4 f1 = *(const float4*)(frow + kb + 4);
      float fv[8] = {f0.x, f0.y, f0.z, f0.w, f1.x, f1.y, f1.z, f1.w};
#pragma unroll
      for (int j = 0; j < 8; ++j) {
        unsigned short h = bf16_rne(fv[j]);
        a_hi[ks].u[run * 8 + j] = h;
        a_lo[ks].u[run * 8 + j] = bf16_rne(fv[j] - bf16_to_f32(h));
      }
    }
  }

  float run_val[8]; int run_n0[8];
#pragma unroll
  for (int r = 0; r < 8; ++r) { run_val[r] = -3.4e38f; run_n0[r] = 0; }

  for (int chunk = 0; chunk < KCODES / NC; ++chunk) {
    __syncthreads();
    for (int e = tid; e < NC * (DIMS / 4); e += 256) {
      const int row = e >> 5;
      const int c4  = (e & 31) * 4;
      float4 v = *(const float4*)(codes + (size_t)(chunk * NC + row) * DIMS + c4);
      float cv[4] = {v.x, v.y, v.z, v.w};
      const int base = row * LDSW + c4;
#pragma unroll
      for (int j = 0; j < 4; ++j) {
        unsigned short h = bf16_rne(cv[j]);
        sBhi[base + j] = h;
        sBlo[base + j] = bf16_rne(cv[j] - bf16_to_f32(h));
      }
    }
    __syncthreads();

#pragma unroll
    for (int sub = 0; sub < NC / 16; ++sub) {
      const int n0 = chunk * NC + sub * 16;
      v8f acc = {};
#pragma unroll
      for (int ks = 0; ks < 4; ++ks) {
        const int cr   = sub * 16 + l16;
        const int koff = ks * 32 + half * 8;
        Frag16 b_hi, b_lo;
        const unsigned short* ph = &sBhi[cr * LDSW + koff];
        b_hi.q[0] = *(const uint4*)(ph);
        b_hi.q[1] = *(const uint4*)(ph + 16);
        const unsigned short* pl = &sBlo[cr * LDSW + koff];
        b_lo.q[0] = *(const uint4*)(pl);
        b_lo.q[1] = *(const uint4*)(pl + 16);
        acc = __builtin_amdgcn_wmma_f32_16x16x32_bf16(false, a_hi[ks].v, false, b_hi.v,
                                                      (short)0, acc, false, false);
        acc = __builtin_amdgcn_wmma_f32_16x16x32_bf16(false, a_hi[ks].v, false, b_lo.v,
                                                      (short)0, acc, false, false);
        acc = __builtin_amdgcn_wmma_f32_16x16x32_bf16(false, a_lo[ks].v, false, b_hi.v,
                                                      (short)0, acc, false, false);
      }
      const float bias = csq[n0 + l16];
#pragma unroll
      for (int r = 0; r < 8; ++r) {
        float s = acc[r] - bias;
        if (s > run_val[r]) { run_val[r] = s; run_n0[r] = n0; }
      }
    }
  }

#pragma unroll
  for (int r = 0; r < 8; ++r) {
    float v = run_val[r];
    int   i = run_n0[r] + l16;
#pragma unroll
    for (int off = 8; off > 0; off >>= 1) {
      float ov = __shfl_xor(v, off, 16);
      int   oi = __shfl_xor(i, off, 16);
      if (ov > v || (ov == v && oi < i)) { v = ov; i = oi; }
    }
    if (l16 == 0) sIdx[wave * 16 + half * 8 + r] = i;
  }
  __syncthreads();

  for (int e = tid; e < MTILE * (DIMS / 4); e += 256) {
    const int row = e >> 5;
    const int c4  = e & 31;
    const int k   = sIdx[row];
    float4 v = ((const float4*)(codes + (size_t)k * DIMS))[c4];
    ((float4*)(out + ((size_t)(blockIdx.x * MTILE + row)) * DIMS))[c4] = v;
  }
}

extern "C" void kernel_launch(void* const* d_in, const int* in_sizes, int n_in,
                              void* d_out, int out_size, void* d_ws, size_t ws_size,
                              hipStream_t stream) {
  const float* features = (const float*)d_in[0];
  const float* codes    = (const float*)d_in[1];
  float* out = (float*)d_out;
  const int N = in_sizes[0] / DIMS;          // 131072

  float* csq = (float*)d_ws;                 // 8 KB
  csq_half_kernel<<<(KCODES + 255) / 256, 256, 0, stream>>>(codes, csq);

  const size_t need = 8192 + (size_t)KCODES * DIMS * 2 * sizeof(unsigned short);
  if (ws_size >= need) {
    unsigned short* chi = (unsigned short*)((char*)d_ws + 8192);
    unsigned short* clo = chi + (size_t)KCODES * DIMS;
    split_codes_kernel<<<(KCODES * DIMS / 4 + 255) / 256, 256, 0, stream>>>(codes, chi, clo);
    vq_main_kernel<<<N / MTILE, 256, 0, stream>>>(features, codes, chi, clo, csq, out);
  } else {
    vq_fallback_kernel<<<N / MTILE, 256, 0, stream>>>(features, codes, csq, out);
  }
}
